// GraphMetricLearning_48996986913408
// MI455X (gfx1250) — matrix operand order repaired and checked by reading, not compiled, not run on hardware
//
#include <hip/hip_runtime.h>
#include <hip/hip_bf16.h>
#include <math.h>

// Problem constants (match reference)
#define BB    16
#define NN    512
#define INDIM 32
#define TT    12
#define DD    64
#define HH    4
#define KTOT  256          // H * D, concatenated-head K dimension
#define LDSP  264          // padded row stride in halves (528 B = 33*16B, banks spread)

typedef _Float16 v16h __attribute__((ext_vector_type(16)));
typedef _Float16 v8h  __attribute__((ext_vector_type(8)));
typedef float    v8f  __attribute__((ext_vector_type(8)));
typedef int      v4i  __attribute__((ext_vector_type(4)));

#if defined(__has_builtin)
#if __has_builtin(__builtin_amdgcn_global_load_async_to_lds_b128) && \
    __has_builtin(__builtin_amdgcn_s_wait_asynccnt)
#define HAS_ASYNC_LDS 1
#endif
#endif

union U16 { v16h v; v8h h[2]; };

__device__ __forceinline__ v16h ld_frag(const _Float16* lo, const _Float16* hi) {
    U16 u;
    u.h[0] = *(const v8h*)lo;
    u.h[1] = *(const v8h*)hi;
    return u.v;
}

// 16-byte global -> LDS copy; async (no VGPR round-trip) when available.
__device__ __forceinline__ void copy16(const _Float16* g, _Float16* l) {
#ifdef HAS_ASYNC_LDS
    __builtin_amdgcn_global_load_async_to_lds_b128(
        (__attribute__((address_space(1))) v4i*)const_cast<_Float16*>(g),
        (__attribute__((address_space(3))) v4i*)l,
        /*offset=*/0, /*cpol=*/0);
#else
    *(v8h*)l = *(const v8h*)g;
#endif
}

// ---------------------------------------------------------------------------
// Kernel 1: h = x@Wp + bp; per (head,t): scale by weights, L2-normalize over D,
// write f16 Xn[(b*T+t)][n][h*64+d]  (row-major, K = 256)
// ---------------------------------------------------------------------------
__global__ void __launch_bounds__(256)
prep_kernel(const float* __restrict__ x, const float* __restrict__ Wp,
            const float* __restrict__ bp, const float* __restrict__ wts,
            _Float16* __restrict__ Xn)
{
    __shared__ float Wsh[INDIM * DD];
    __shared__ float bsh[DD];
    const int tid = threadIdx.x;
    for (int i = tid; i < INDIM * DD; i += 256) Wsh[i] = Wp[i];
    if (tid < DD) bsh[tid] = bp[tid];
    __syncthreads();

    const int row = blockIdx.x * 256 + tid;       // 0 .. B*N-1
    const int b = row >> 9;                        // /512
    const int n = row & 511;

    const float* xr = x + (size_t)row * INDIM;
    float xv[INDIM];
    #pragma unroll
    for (int k = 0; k < INDIM; ++k) xv[k] = xr[k];

    float h[DD];
    #pragma unroll
    for (int d = 0; d < DD; ++d) {
        float acc = bsh[d];
        #pragma unroll
        for (int k = 0; k < INDIM; ++k) acc = fmaf(xv[k], Wsh[k * DD + d], acc);
        h[d] = acc;
    }

    for (int hh = 0; hh < HH; ++hh) {
        for (int t = 0; t < TT; ++t) {
            const float* w = wts + ((size_t)(hh * TT + t)) * DD;  // uniform -> scalar loads
            float ss = 0.f;
            #pragma unroll
            for (int d = 0; d < DD; ++d) {
                float p = h[d] * w[d];
                ss = fmaf(p, p, ss);
            }
            const float inv = 1.0f / fmaxf(sqrtf(ss), 1e-12f);
            _Float16* dst = Xn + (((size_t)(b * TT + t) * NN + n) * KTOT + hh * DD);
            #pragma unroll
            for (int c = 0; c < DD / 8; ++c) {
                v8h pk;
                #pragma unroll
                for (int j = 0; j < 8; ++j)
                    pk[j] = (_Float16)(h[c * 8 + j] * w[c * 8 + j] * inv);
                *(v8h*)(dst + c * 8) = pk;        // global_store_b128
            }
        }
    }
}

// ---------------------------------------------------------------------------
// Kernel 2: batched Gram  out[bt] = Xn[bt] * Xn[bt]^T / H   (512x512, K=256)
// grid: (4 col macro, 4 row macro, 192 bt); block 256 = 8 wave32
// macro-tile 128x128 staged in LDS via async global->LDS; each wave owns a
// 2x4 grid of 16x16 tiles; K-loop double-buffers fragments across steps.
// ---------------------------------------------------------------------------
__global__ void __launch_bounds__(256)
gram_kernel(const _Float16* __restrict__ Xn, float* __restrict__ out)
{
    extern __shared__ __align__(16) _Float16 smem[];   // 2 * 128 * LDSP halves
    _Float16* As = smem;                 // rows  [rowBase, rowBase+128)
    _Float16* Bs = smem + 128 * LDSP;    // rows  [colBase, colBase+128)

    const int tid     = threadIdx.x;
    const int bt      = blockIdx.z;
    const int rowBase = blockIdx.y * 128;
    const int colBase = blockIdx.x * 128;
    const _Float16* Xg = Xn + (size_t)bt * NN * KTOT;

    // cooperative fill: 2 x 128 rows x 512B, 16B chunks (async -> LDS)
    for (int c = tid; c < 4096; c += 256) {
        const int r = c >> 5;
        const int o = (c & 31) * 8;      // halves
        copy16(Xg + (size_t)(rowBase + r) * KTOT + o, As + r * LDSP + o);
        copy16(Xg + (size_t)(colBase + r) * KTOT + o, Bs + r * LDSP + o);
    }
#ifdef HAS_ASYNC_LDS
    __builtin_amdgcn_s_wait_asynccnt(0);
#endif
    __syncthreads();

    const int lane = tid & 31;
    const int wave = tid >> 5;
    const int wrow = wave >> 1;          // 0..3 -> 2 row-tiles each
    const int wcol = wave & 1;           // 0..1 -> 4 col-tiles each
    const int l16  = lane & 15;
    const int hi   = lane >> 4;          // half-wave select

    const _Float16* aBase = As + (size_t)(wrow * 32 + l16) * LDSP + hi * 16;
    const _Float16* bBase = Bs + (size_t)(wcol * 64 + l16) * LDSP + hi * 16;

    v8f acc[2][4];
    const v8f vzero = {0.f, 0.f, 0.f, 0.f, 0.f, 0.f, 0.f, 0.f};
    #pragma unroll
    for (int a = 0; a < 2; ++a)
        #pragma unroll
        for (int c = 0; c < 4; ++c) acc[a][c] = vzero;

    v16h af[2][2], bf[2][4];

    // prologue: fragments for K-step 0
    #pragma unroll
    for (int a = 0; a < 2; ++a) {
        const _Float16* p = aBase + (size_t)(a * 16) * LDSP;
        af[0][a] = ld_frag(p, p + 8);
    }
    #pragma unroll
    for (int c = 0; c < 4; ++c) {
        const _Float16* p = bBase + (size_t)(c * 16) * LDSP;
        bf[0][c] = ld_frag(p, p + 8);
    }

    #pragma unroll
    for (int ks = 0; ks < 8; ++ks) {
        const int cur = ks & 1;
        const int nxt = cur ^ 1;
        if (ks < 7) {                    // prefetch next K-step's fragments
            const int kk = (ks + 1) * 32;
            #pragma unroll
            for (int a = 0; a < 2; ++a) {
                const _Float16* p = aBase + (size_t)(a * 16) * LDSP + kk;
                af[nxt][a] = ld_frag(p, p + 8);
            }
            #pragma unroll
            for (int c = 0; c < 4; ++c) {
                const _Float16* p = bBase + (size_t)(c * 16) * LDSP + kk;
                bf[nxt][c] = ld_frag(p, p + 8);
            }
        }
        #pragma unroll
        for (int a = 0; a < 2; ++a)
            #pragma unroll
            for (int c = 0; c < 4; ++c)
                acc[a][c] = __builtin_amdgcn_wmma_f32_16x16x32_f16(
                    false, af[cur][a], false, bf[cur][c], (short)0, acc[a][c], false, false);
    }

    // store D (16x16 f32 layout): lanes 0-15 -> M=0..7 rows, lanes 16-31 -> M=8..15
    float* ob = out + (size_t)bt * NN * NN;
    #pragma unroll
    for (int a = 0; a < 2; ++a) {
        const int r0 = rowBase + (wrow * 2 + a) * 16 + hi * 8;
        #pragma unroll
        for (int c = 0; c < 4; ++c) {
            const int col = colBase + (wcol * 4 + c) * 16 + l16;
            #pragma unroll
            for (int r = 0; r < 8; ++r)
                ob[(size_t)(r0 + r) * NN + col] = acc[a][c][r] * 0.25f;  // / H
        }
    }
}

// ---------------------------------------------------------------------------
extern "C" void kernel_launch(void* const* d_in, const int* in_sizes, int n_in,
                              void* d_out, int out_size, void* d_ws, size_t ws_size,
                              hipStream_t stream)
{
    const float* x   = (const float*)d_in[0];   // (B, N, 32)
    const float* Wp  = (const float*)d_in[1];   // (32, 64)
    const float* bp  = (const float*)d_in[2];   // (64)
    const float* wts = (const float*)d_in[3];   // (H, T, 1, 64)
    _Float16* Xn = (_Float16*)d_ws;             // (B*T, N, 256) f16 = 50,331,648 B
    float* out = (float*)d_out;                 // (B, T, N, N) f32

    (void)in_sizes; (void)n_in; (void)out_size; (void)ws_size;

    prep_kernel<<<dim3((BB * NN) / 256), dim3(256), 0, stream>>>(x, Wp, bp, wts, Xn);

    const size_t lds_bytes = (size_t)2 * 128 * LDSP * sizeof(_Float16);  // 135,168 B
    gram_kernel<<<dim3(4, 4, BB * TT), dim3(256), lds_bytes, stream>>>(Xn, out);
}
